// Transformer_GNN_58239756534356
// MI455X (gfx1250) — matrix operand order repaired and checked by reading, not compiled, not run on hardware
//
#include <hip/hip_runtime.h>
#include <hip/hip_bf16.h>
#include <math.h>

typedef __attribute__((ext_vector_type(16))) _Float16 v16h;
typedef __attribute__((ext_vector_type(8)))  float    v8f;

// ---------------------------------------------------------------------------
// Helpers
// ---------------------------------------------------------------------------
__device__ __forceinline__ void atomicMaxF(float* addr, float val) {
  // standard monotonic reinterpretation trick: works with -inf init
  if (val >= 0.0f) atomicMax((int*)addr,          __float_as_int(val));
  else             atomicMin((unsigned int*)addr, __float_as_uint(val));
}

// ---------------------------------------------------------------------------
// One 16x16 output tile of  C = A[M x Din] * W[Din x 96] + bias,  f32 in/out.
// Uses V_WMMA_F32_16X16X32_F16 with an f16 hi/lo split (3 WMMAs per k-step)
// for ~fp32 accuracy. Fragment layouts per CDNA5 ISA 7.12.2 (wave32).
// ---------------------------------------------------------------------------
__device__ __forceinline__ void gemm_tile_16x16(
    const float* __restrict__ A, int lda,
    const float* __restrict__ W,        // [Din x 96] row-major
    const float* __restrict__ bias,     // [96]
    float* __restrict__ C, int ldc,
    int m0, int nbase, int Din, int lane) {
  const int g   = lane >> 4;            // lane group (0: lanes 0-15, 1: 16-31)
  const int l15 = lane & 15;
  v8f acc = {0.f, 0.f, 0.f, 0.f, 0.f, 0.f, 0.f, 0.f};

  for (int kb = 0; kb < Din; kb += 32) {
    // A fragment 16x32 f16: lane holds row M=l15; halves h=0..7 -> K=kb+8g+h,
    // halves h=8..15 -> K=kb+16+8g+(h-8). Two contiguous 8-float runs.
    v16h ah, al;
    {
      const float* r0 = A + (size_t)(m0 + l15) * lda + kb + 8 * g;
      const float* r1 = r0 + 16;
#pragma unroll
      for (int i = 0; i < 8; ++i) {
        float x = r0[i];
        _Float16 h = (_Float16)x;
        ah[i] = h;
        al[i] = (_Float16)(x - (float)h);
      }
#pragma unroll
      for (int i = 0; i < 8; ++i) {
        float x = r1[i];
        _Float16 h = (_Float16)x;
        ah[8 + i] = h;
        al[8 + i] = (_Float16)(x - (float)h);
      }
    }
    // B fragment 32x16 f16: lane holds col N=nbase+l15; halves h=0..7 -> K=kb+8g+h,
    // halves h=8..15 -> K=kb+16+8g+(h-8).
    v16h bh, bl;
    {
      const float* c0 = W + (size_t)(kb + 8 * g) * 96 + nbase + l15;
#pragma unroll
      for (int i = 0; i < 16; ++i) {
        float x = c0[(size_t)(i < 8 ? i : i + 8) * 96];
        _Float16 h = (_Float16)x;
        bh[i] = h;
        bl[i] = (_Float16)(x - (float)h);
      }
    }
    // C += Ahi*Bhi + Ahi*Blo + Alo*Bhi   (drop lo*lo ~ 2^-22)
    acc = __builtin_amdgcn_wmma_f32_16x16x32_f16(false, ah, false, bh,
                                                 (short)0, acc, false, false);
    acc = __builtin_amdgcn_wmma_f32_16x16x32_f16(false, ah, false, bl,
                                                 (short)0, acc, false, false);
    acc = __builtin_amdgcn_wmma_f32_16x16x32_f16(false, al, false, bh,
                                                 (short)0, acc, false, false);
  }

  const float b = bias[nbase + l15];
  // D layout: VGPR r -> row m0 + r + 8*g, col nbase + l15
#pragma unroll
  for (int r = 0; r < 8; ++r)
    C[(size_t)(m0 + r + 8 * g) * ldc + nbase + l15] = acc[r] + b;
}

// ---------------------------------------------------------------------------
// Fused multi-projection GEMM. Each wave (32 threads) computes one 16x16 tile.
// task = proj*6 + column-tile (96 cols = 6 tiles). Wave-uniform branching so
// EXEC is all-ones around WMMA.
// ---------------------------------------------------------------------------
__global__ void __launch_bounds__(256)
gnn_gemm_kernel(const float* __restrict__ A, int Din, int M,
                const float* w0, const float* b0, float* o0,
                const float* w1, const float* b1, float* o1,
                const float* w2, const float* b2, float* o2,
                const float* w3, const float* b3, float* o3,
                int ntasks) {
  const int lane = threadIdx.x;                       // 0..31 (wave32)
  const int task = blockIdx.y * blockDim.y + threadIdx.y;
  if (task >= ntasks) return;                         // wave-uniform
  const int m0 = blockIdx.x * 16;
  if (m0 + 16 > M) return;                            // wave-uniform
  const int proj = task / 6;
  const int nt   = task % 6;
  const float* W; const float* B; float* O;
  if      (proj == 0) { W = w0; B = b0; O = o0; }
  else if (proj == 1) { W = w1; B = b1; O = o1; }
  else if (proj == 2) { W = w2; B = b2; O = o2; }
  else                { W = w3; B = b3; O = o3; }
  gemm_tile_16x16(A, Din, W, B, O, 96, m0, nt * 16, Din, lane);
}

// ---------------------------------------------------------------------------
// Edge-phase kernels (segment softmax over destination nodes)
// ---------------------------------------------------------------------------
__global__ void init_md_kernel(float* __restrict__ m, float* __restrict__ d,
                               int n) {
  int t = blockIdx.x * blockDim.x + threadIdx.x;
  if (t >= n) return;
  m[t] = -INFINITY;
  d[t] = 0.0f;
}

// score[e,h] = (q[dst,h,:] . k[src,h,:]) / sqrt(12);  running atomic max
__global__ void edge_score_kernel(const int* __restrict__ ei,
                                  const float* __restrict__ q,
                                  const float* __restrict__ k,
                                  float* __restrict__ score,
                                  float* __restrict__ mbuf, int E) {
  int t = blockIdx.x * blockDim.x + threadIdx.x;
  if (t >= E * 8) return;
  const int e = t >> 3, h = t & 7;
  const int s = ei[e];
  const int d = ei[E + e];
  const float* qp = q + (size_t)d * 96 + h * 12;
  const float* kp = k + (size_t)s * 96 + h * 12;
  __builtin_prefetch(k + (size_t)s * 96, 0, 1);   // global_prefetch_b8
  float acc = 0.0f;
#pragma unroll
  for (int i = 0; i < 12; ++i) acc += qp[i] * kp[i];
  acc *= 0.2886751345948129f;                      // 1/sqrt(12)
  score[t] = acc;
  atomicMaxF(&mbuf[(size_t)d * 8 + h], acc);
}

// e = exp(score - m[dst]);  denom[dst] += e;  score <- e
__global__ void edge_exp_kernel(const int* __restrict__ ei,
                                float* __restrict__ score,
                                const float* __restrict__ mbuf,
                                float* __restrict__ dbuf, int E) {
  int t = blockIdx.x * blockDim.x + threadIdx.x;
  if (t >= E * 8) return;
  const int e = t >> 3, h = t & 7;
  const int d = ei[E + e];
  const float m  = mbuf[(size_t)d * 8 + h];        // finite: d has >=1 edge
  const float ex = expf(score[t] - m);
  score[t] = ex;
  atomicAdd(&dbuf[(size_t)d * 8 + h], ex);
}

// out[dst,h,:] += (e / (denom[dst,h]+EPS)) * v[src,h,:]
__global__ void edge_msg_kernel(const int* __restrict__ ei,
                                const float* __restrict__ score,
                                const float* __restrict__ dbuf,
                                const float* __restrict__ v,
                                float* __restrict__ out, int E) {
  int t = blockIdx.x * blockDim.x + threadIdx.x;
  if (t >= E * 8) return;
  const int e = t >> 3, h = t & 7;
  const int s = ei[e];
  const int d = ei[E + e];
  const float alpha = score[t] / (dbuf[(size_t)d * 8 + h] + 1e-16f);
  const float* vp = v + (size_t)s * 96 + h * 12;
  float* op = out + (size_t)d * 96 + h * 12;
#pragma unroll
  for (int i = 0; i < 12; ++i) atomicAdd(&op[i], alpha * vp[i]);
}

// exact GELU: 0.5*x*(1+erf(x/sqrt(2)))
__global__ void gelu_kernel(const float* __restrict__ in,
                            float* __restrict__ out, int n) {
  int t = blockIdx.x * blockDim.x + threadIdx.x;
  if (t >= n) return;
  const float x = in[t];
  out[t] = 0.5f * x * (1.0f + erff(x * 0.7071067811865476f));
}

// ---------------------------------------------------------------------------
// Host-side orchestration
// ---------------------------------------------------------------------------
extern "C" void kernel_launch(void* const* d_in, const int* in_sizes, int n_in,
                              void* d_out, int out_size, void* d_ws,
                              size_t ws_size, hipStream_t stream) {
  (void)n_in; (void)out_size; (void)ws_size;
  const int N = in_sizes[0] / 64;        // x: [N,64]
  const int E = in_sizes[1] / 2;         // edge_index: [2,E]

  const float* x  = (const float*)d_in[0];
  const int*   ei = (const int*)d_in[1];
  const float* P[4][8];
  for (int l = 0; l < 4; ++l)
    for (int j = 0; j < 8; ++j)
      P[l][j] = (const float*)d_in[2 + l * 8 + j];   // wq,bq,wk,bk,wv,bv,ws,bs
  const float* fw = (const float*)d_in[34];
  const float* fb = (const float*)d_in[35];
  float* dout = (float*)d_out;

  // workspace layout (floats)
  float* w    = (float*)d_ws;
  float* q    = w;                              // N*96
  float* kbf  = q    + (size_t)N * 96;          // N*96
  float* vbf  = kbf  + (size_t)N * 96;          // N*96
  float* agg  = vbf  + (size_t)N * 96;          // N*96  (skip + messages)
  float* hbuf = agg  + (size_t)N * 96;          // N*96  (gelu output)
  float* sc   = hbuf + (size_t)N * 96;          // E*8
  float* mb   = sc   + (size_t)E * 8;           // N*8
  float* db   = mb   + (size_t)N * 8;           // N*8

  const dim3 gemm_blk(32, 8);
  const int  row_tiles = (N + 15) / 16;         // N=50000 -> 3125 exact
  const int  eth = (E * 8 + 255) / 256;
  const int  nth = (N * 8 + 255) / 256;
  const int  fth = (N * 96 + 255) / 256;

  for (int l = 0; l < 4; ++l) {
    const float* A  = (l == 0) ? x : hbuf;
    const int   Din = (l == 0) ? 64 : 96;
    // q,k,v projections + skip projection (written straight into agg)
    gnn_gemm_kernel<<<dim3(row_tiles, 3), gemm_blk, 0, stream>>>(
        A, Din, N,
        P[l][0], P[l][1], q,
        P[l][2], P[l][3], kbf,
        P[l][4], P[l][5], vbf,
        P[l][6], P[l][7], agg, 24);
    init_md_kernel<<<nth, 256, 0, stream>>>(mb, db, N * 8);
    edge_score_kernel<<<eth, 256, 0, stream>>>(ei, q, kbf, sc, mb, E);
    edge_exp_kernel<<<eth, 256, 0, stream>>>(ei, sc, mb, db, E);
    edge_msg_kernel<<<eth, 256, 0, stream>>>(ei, sc, db, vbf, agg, E);
    if (l < 3)
      gelu_kernel<<<fth, 256, 0, stream>>>(agg, hbuf, N * 96);
  }

  // final linear: d_out = h3 @ final_w + final_b   (only 6 tasks used)
  gnn_gemm_kernel<<<dim3(row_tiles, 1), gemm_blk, 0, stream>>>(
      agg, 96, N,
      fw, fb, dout, fw, fb, dout, fw, fb, dout, fw, fb, dout, 6);
}
